// MORRMHA_33389075759465
// MI455X (gfx1250) — matrix operand order repaired and checked by reading, not compiled, not run on hardware
//
#include <hip/hip_runtime.h>
#include <hip/hip_bf16.h>
#include <stdint.h>

// ---------------------------------------------------------------------------
// Problem constants (fixed by reference): B=2, N=1024, D=1024, H=16, dh=64,
// circulant block CK=8, p=q=128.  M = B*N = 2048 rows for all GEMMs.
// ---------------------------------------------------------------------------
constexpr int D     = 1024;
constexpr int NSEQ  = 1024;
constexpr int BATCH = 2;
constexpr int MROWS = BATCH * NSEQ;   // 2048
constexpr int HEADS = 16;
constexpr int DH    = 64;
constexpr int CK    = 8;
constexpr int PQ    = D / CK;         // 128

typedef __attribute__((ext_vector_type(16))) __bf16 v16bf;
typedef __attribute__((ext_vector_type(8)))  __bf16 v8bf;
typedef __attribute__((ext_vector_type(8)))  float  v8f;

// v_wmma_f32_16x16x32_bf16 : D(16x16 f32) = A(16x32 bf16) * B(32x16 bf16) + C
__device__ __forceinline__ v8f wmma_bf16(v16bf a, v16bf b, v8f c) {
  return __builtin_amdgcn_wmma_f32_16x16x32_bf16(
      /*neg_a=*/false, a, /*neg_b=*/false, b,
      /*c_mod=*/(short)0, c, /*reuse_a=*/false, /*reuse_b=*/false);
}

// A-fragment loader (row-major source).
// lane l: row = l&15, kbase = (l>>4)*16 ; elems 0..15 -> k=kbase..kbase+15.
// `p` must already point at
// &A[row*ld + kbase] for the current 32-wide k-slice.
__device__ __forceinline__ v16bf load_afrag(const __bf16* p) {
  v8bf lo = *(const v8bf*)(p);
  v8bf hi = *(const v8bf*)(p + 8);
  return __builtin_shufflevector(lo, hi, 0, 1, 2, 3, 4, 5, 6, 7,
                                          8, 9, 10, 11, 12, 13, 14, 15);
}

// CDNA5 async copy: 16B global -> LDS, tracked by ASYNCcnt (GV mode).
// Generic pointers to __shared__ carry the LDS byte offset in their low 32
// bits (flat aperture rule, ISA 10.2), so truncation gives the DS address.
__device__ __forceinline__ void async_copy_b128(uint32_t lds_addr,
                                                uint64_t gaddr) {
  asm volatile("global_load_async_to_lds_b128 %0, %1, off"
               :: "v"(lds_addr), "v"(gaddr)
               : "memory");
}

// ---------------------------------------------------------------------------
// Kernel 1: expand block-circulant weights [128,128,8] f32 into a dense
// bf16 matrix Wd[d_out, d_in] with Wd[p*8+i, q*8+j] = w[p,q,(i-j) mod 8].
// ---------------------------------------------------------------------------
__global__ void __launch_bounds__(256)
expand_circulant_kernel(const float* __restrict__ w, __bf16* __restrict__ Wd) {
  const int idx  = blockIdx.x * 256 + threadIdx.x;   // 0 .. D*D-1
  const int dout = idx >> 10;
  const int din  = idx & (D - 1);
  const int p = dout >> 3, i = dout & 7;
  const int q = din  >> 3, j = din  & 7;
  Wd[idx] = (__bf16)w[(p * PQ + q) * CK + ((i - j) & 7)];
}

// ---------------------------------------------------------------------------
// Kernel 2: f32 -> bf16 convert (activations).
// ---------------------------------------------------------------------------
__global__ void __launch_bounds__(256)
f32_to_bf16_kernel(const float* __restrict__ in, __bf16* __restrict__ out) {
  const int idx = blockIdx.x * 256 + threadIdx.x;
  out[idx] = (__bf16)in[idx];
}

// ---------------------------------------------------------------------------
// Kernel 3: bf16 GEMM   C[row,col] = sum_k A[row,k] * Bm[col,k] + bias[col]
//   A  : [MROWS, D] bf16 row-major        (direct global_load_b128 fragments)
//   Bm : [D, D]     bf16 row-major (B^T storage)
// Block = 256 threads = 8 waves, block tile 128(M) x 64(N),
// wave tile 32x32 (2x2 WMMA accumulators), k-loop in steps of 32.
//
// B panel pipeline (CDNA5 async path): per 32-k step the block's 64x32 bf16
// B slice (4KB) is staged into LDS with global_load_async_to_lds_b128
// (one 16B async copy per thread), double-buffered so the copy for step k+32
// overlaps the WMMAs of step k; consumers sync with s_wait_asynccnt +
// workgroup barrier and read contiguous 32B B-fragments from LDS.
//
// Epilogue modes:
//   1: bf16 head-major      out[b,h,n,d]        (Q, K)
//   2: bf16 head-transposed out[b,h,d,n]        (V^T)
//   3: f32  row-major       out[row, col]       (final output)
// ---------------------------------------------------------------------------
__global__ void __launch_bounds__(256)
gemm_bf16_kernel(const __bf16* __restrict__ A, const __bf16* __restrict__ Bm,
                 const float* __restrict__ bias, void* __restrict__ outp,
                 int mode) {
  __shared__ alignas(128) __bf16 bpanel[2][64 * 32];   // 2 x 4KB double buffer

  const int tid   = threadIdx.x;
  const int lane  = tid & 31;
  const int wave  = tid >> 5;
  const int colid = lane & 15;
  const int half  = lane >> 4;
  const int tm  = blockIdx.x * 128 + (wave & 3) * 32;
  const int tnb = blockIdx.y * 64;

  // --- async B-panel producer: thread tid copies 16B of the 64x32 slice ---
  const int pcol = tid >> 2;                     // 0..63  (panel column)
  const int pseg = tid & 3;                      // 0..3   (8 bf16 each)
  const uint64_t gsrc =
      (uint64_t)(uintptr_t)(Bm + (size_t)(tnb + pcol) * D + pseg * 8);
  const uint32_t ldst0 =
      (uint32_t)(uintptr_t)(&bpanel[0][pcol * 32 + pseg * 8]);
  const uint32_t ldst1 =
      (uint32_t)(uintptr_t)(&bpanel[1][pcol * 32 + pseg * 8]);

  async_copy_b128(ldst0, gsrc);                  // preload k = 0 slice

  v8f c00 = {}, c01 = {}, c10 = {}, c11 = {};
  const __bf16* a0p = A + (size_t)(tm + colid) * D + half * 16;
  const __bf16* a1p = a0p + (size_t)16 * D;
  // B-fragment read base in LDS: col_local = waveN*32 + colid, k = half*16+e
  const __bf16* bfrag0 =
      &bpanel[0][((wave >> 2) * 32 + colid) * 32 + half * 16];

  for (int k = 0; k < D; k += 32) {
    const int cur = (k >> 5) & 1;
    const bool pre = (k + 32) < D;
    if (pre) {
      const uint32_t ld = cur ? ldst0 : ldst1;
      async_copy_b128(ld, gsrc + (uint64_t)(k + 32) * sizeof(__bf16));
      asm volatile("s_wait_asynccnt 0x1" ::: "memory");
    } else {
      asm volatile("s_wait_asynccnt 0x0" ::: "memory");
    }
    __syncthreads();                             // panel[cur] ready block-wide

    v16bf af0 = load_afrag(a0p + k);
    v16bf af1 = load_afrag(a1p + k);
    const __bf16* bp = bfrag0 + (size_t)cur * (64 * 32);
    v16bf bf0 = *(const v16bf*)(bp);             // cols tn .. tn+15
    v16bf bf1 = *(const v16bf*)(bp + 16 * 32);   // cols tn+16 .. tn+31
    c00 = wmma_bf16(af0, bf0, c00);
    c01 = wmma_bf16(af0, bf1, c01);
    c10 = wmma_bf16(af1, bf0, c10);
    c11 = wmma_bf16(af1, bf1, c11);

    __syncthreads();                             // reads done before rewrite
  }

  const int tn = tnb + (wave >> 2) * 32;
  auto store_tile = [&](const v8f& c, int rbase, int cbase) {
    const int col = cbase + colid;
    const float bv = bias[col];
#pragma unroll
    for (int r = 0; r < 8; ++r) {
      const int row = rbase + r + 8 * half;
      const float v = c[r] + bv;
      if (mode == 1) {
        const int bb = row >> 10, n = row & (NSEQ - 1);
        const int hh = col >> 6,  dd = col & (DH - 1);
        ((__bf16*)outp)[((size_t)((bb * HEADS + hh) * NSEQ + n)) * DH + dd] =
            (__bf16)v;
      } else if (mode == 2) {
        const int bb = row >> 10, n = row & (NSEQ - 1);
        const int hh = col >> 6,  dd = col & (DH - 1);
        ((__bf16*)outp)[((size_t)((bb * HEADS + hh) * DH + dd)) * NSEQ + n] =
            (__bf16)v;
      } else {
        ((float*)outp)[(size_t)row * D + col] = v;
      }
    }
  };
  store_tile(c00, tm,      tn);
  store_tile(c01, tm,      tn + 16);
  store_tile(c10, tm + 16, tn);
  store_tile(c11, tm + 16, tn + 16);
}

// ---------------------------------------------------------------------------
// Kernel 4: causal flash attention, bf16 WMMA, f32 online softmax.
//   Q,K : [B,H,N,DH] bf16      V^T : [B,H,DH,N] bf16
//   Ao  : [B,N,H*DH] bf16  (ready as GEMM A for the output projection)
// One wave owns a 16-row Q tile; streams K/V in 32-wide m-chunks.
// Per chunk: 4 WMMA for S (two 16x16 subtiles x 2 k-steps over dh=64),
// online max/sum (butterfly shfl over 16-lane halves), P staged through LDS
// (C-layout -> A-layout), 4 WMMA for O += P*V.
// ---------------------------------------------------------------------------
__global__ void __launch_bounds__(128)
attn_kernel(const __bf16* __restrict__ Q, const __bf16* __restrict__ Kmat,
            const __bf16* __restrict__ Vt, __bf16* __restrict__ Ao) {
  __shared__ alignas(64) __bf16 pst[4][16 * 32];   // per-wave P staging

  const int lane  = threadIdx.x & 31;
  const int wave  = threadIdx.x >> 5;
  const int colid = lane & 15;
  const int half  = lane >> 4;

  const int gid = blockIdx.x * 4 + wave;           // 0 .. 2047
  const int qt  = gid & 63;
  const int h   = (gid >> 6) & (HEADS - 1);
  const int b   = gid >> 10;
  const int qbase = qt * 16;

  const __bf16* Qb = Q    + (size_t)((b * HEADS + h) * NSEQ) * DH;
  const __bf16* Kb = Kmat + (size_t)((b * HEADS + h) * NSEQ) * DH;
  const __bf16* Vb = Vt   + (size_t)((b * HEADS + h) * DH) * NSEQ;

  // Q A-fragments for d=0..31 and d=32..63
  const __bf16* qrow = Qb + (size_t)(qbase + colid) * DH + half * 16;
  const v16bf qf0 = load_afrag(qrow);
  const v16bf qf1 = load_afrag(qrow + 32);

  v8f o0 = {}, o1 = {}, o2 = {}, o3 = {};
  float mrun[8], lrun[8];
#pragma unroll
  for (int r = 0; r < 8; ++r) { mrun[r] = -1e30f; lrun[r] = 0.0f; }

  const int nchunk = qt / 2 + 1;                   // causal: m <= qbase+15
  for (int c = 0; c < nchunk; ++c) {
    const int m0 = c * 32;
    if (c + 1 < nchunk) {                          // hint next K/V chunk
      __builtin_prefetch(Kb + (size_t)(m0 + 32) * DH, 0, 3);
      __builtin_prefetch(Vb + m0 + 32, 0, 3);
    }

    // ---- S = Q K^T (two 16-col subtiles) -------------------------------
    v8f s0 = {}, s1 = {};
    {
      const __bf16* kp = Kb + (size_t)(m0 + colid) * DH + half * 16;
      s0 = wmma_bf16(qf0, *(const v16bf*)(kp),      s0);
      s0 = wmma_bf16(qf1, *(const v16bf*)(kp + 32), s0);
    }
    const bool do1 = (m0 + 16) <= (qbase + 15);    // wave-uniform
    if (do1) {
      const __bf16* kp = Kb + (size_t)(m0 + 16 + colid) * DH + half * 16;
      s1 = wmma_bf16(qf0, *(const v16bf*)(kp),      s1);
      s1 = wmma_bf16(qf1, *(const v16bf*)(kp + 32), s1);
    }

    // ---- scale + causal mask ------------------------------------------
#pragma unroll
    for (int r = 0; r < 8; ++r) {
      const int row = qbase + r + 8 * half;
      float v0 = s0[r] * 0.125f;                   // 1/sqrt(dh)
      if (m0 + colid > row) v0 = -1e30f;
      s0[r] = v0;
      float v1 = do1 ? s1[r] * 0.125f : -1e30f;
      if (m0 + 16 + colid > row) v1 = -1e30f;
      s1[r] = v1;
    }

    // ---- row max across this 32-wide chunk ----------------------------
    float tmax[8];
#pragma unroll
    for (int r = 0; r < 8; ++r) tmax[r] = fmaxf(s0[r], s1[r]);
#pragma unroll
    for (int off = 1; off <= 8; off <<= 1)
#pragma unroll
      for (int r = 0; r < 8; ++r)
        tmax[r] = fmaxf(tmax[r], __shfl_xor(tmax[r], off, 32));

    // ---- online softmax update ----------------------------------------
    float corr[8], rs[8];
#pragma unroll
    for (int r = 0; r < 8; ++r) {
      const float mn = fmaxf(mrun[r], tmax[r]);
      corr[r] = __expf(mrun[r] - mn);
      mrun[r] = mn;
      const float p0 = __expf(s0[r] - mn);
      const float p1 = __expf(s1[r] - mn);
      s0[r] = p0;
      s1[r] = p1;
      rs[r] = p0 + p1;
    }
#pragma unroll
    for (int off = 1; off <= 8; off <<= 1)
#pragma unroll
      for (int r = 0; r < 8; ++r) rs[r] += __shfl_xor(rs[r], off, 32);
#pragma unroll
    for (int r = 0; r < 8; ++r) {
      lrun[r] = lrun[r] * corr[r] + rs[r];
      o0[r] *= corr[r]; o1[r] *= corr[r];
      o2[r] *= corr[r]; o3[r] *= corr[r];
    }

    // ---- P: C-layout -> LDS (row-major 16x32 bf16) -> A-fragment ------
    __bf16* ps = pst[wave];
    __builtin_amdgcn_wave_barrier();
#pragma unroll
    for (int r = 0; r < 8; ++r) {
      const int prow = r + 8 * half;
      ps[prow * 32 + colid]      = (__bf16)s0[r];
      ps[prow * 32 + 16 + colid] = (__bf16)s1[r];
    }
    __builtin_amdgcn_wave_barrier();
    asm volatile("s_wait_dscnt 0x0" ::: "memory");

    const __bf16* pp = ps + colid * 32 + half * 16;
    const v16bf pf = load_afrag(pp);

    // ---- O += P * V  (V^T gives contiguous B-fragments) ---------------
    const __bf16* vbase = Vb + (size_t)colid * NSEQ + m0 + half * 16;
    o0 = wmma_bf16(pf, *(const v16bf*)(vbase),                     o0);
    o1 = wmma_bf16(pf, *(const v16bf*)(vbase + (size_t)16 * NSEQ), o1);
    o2 = wmma_bf16(pf, *(const v16bf*)(vbase + (size_t)32 * NSEQ), o2);
    o3 = wmma_bf16(pf, *(const v16bf*)(vbase + (size_t)48 * NSEQ), o3);
  }

  // ---- normalize and store to [B, N, H*DH] bf16 -----------------------
#pragma unroll
  for (int r = 0; r < 8; ++r) {
    const float inv = 1.0f / lrun[r];
    const int row = qbase + r + 8 * half;
    const size_t base = ((size_t)(b * NSEQ + row)) * D + h * DH + colid;
    Ao[base + 0]  = (__bf16)(o0[r] * inv);
    Ao[base + 16] = (__bf16)(o1[r] * inv);
    Ao[base + 32] = (__bf16)(o2[r] * inv);
    Ao[base + 48] = (__bf16)(o3[r] * inv);
  }
}

// ---------------------------------------------------------------------------
// Launch: expand weights, convert x, QKV GEMMs, attention, output GEMM.
// Workspace layout (bytes):
//   [0,2M)   Wq dense bf16      [2M,4M)  Wk      [4M,6M)  Wv     [6M,8M) Wo
//   [8M,12M) x bf16 [2048,1024]
//   [12M,16M) Q [B,H,N,dh]  [16M,20M) K [B,H,N,dh]  [20M,24M) V^T [B,H,dh,N]
//   [24M,28M) attention out bf16 [2048,1024]
// ---------------------------------------------------------------------------
extern "C" void kernel_launch(void* const* d_in, const int* in_sizes, int n_in,
                              void* d_out, int out_size, void* d_ws,
                              size_t ws_size, hipStream_t stream) {
  const float* x  = (const float*)d_in[0];
  // d_in[1] = causal mask (implicit in the kernel; unused)
  const float* wq = (const float*)d_in[2];
  const float* bq = (const float*)d_in[3];
  const float* wk = (const float*)d_in[4];
  const float* bk = (const float*)d_in[5];
  const float* wv = (const float*)d_in[6];
  const float* bv = (const float*)d_in[7];
  const float* wo = (const float*)d_in[8];
  const float* bo = (const float*)d_in[9];

  constexpr size_t MB = 1u << 20;
  char* ws = (char*)d_ws;
  __bf16* Wqd = (__bf16*)(ws + 0 * MB);
  __bf16* Wkd = (__bf16*)(ws + 2 * MB);
  __bf16* Wvd = (__bf16*)(ws + 4 * MB);
  __bf16* Wod = (__bf16*)(ws + 6 * MB);
  __bf16* Xb  = (__bf16*)(ws + 8 * MB);
  __bf16* Qd  = (__bf16*)(ws + 12 * MB);
  __bf16* Kd  = (__bf16*)(ws + 16 * MB);
  __bf16* Vtd = (__bf16*)(ws + 20 * MB);
  __bf16* Aod = (__bf16*)(ws + 24 * MB);

  expand_circulant_kernel<<<(D * D) / 256, 256, 0, stream>>>(wq, Wqd);
  expand_circulant_kernel<<<(D * D) / 256, 256, 0, stream>>>(wk, Wkd);
  expand_circulant_kernel<<<(D * D) / 256, 256, 0, stream>>>(wv, Wvd);
  expand_circulant_kernel<<<(D * D) / 256, 256, 0, stream>>>(wo, Wod);
  f32_to_bf16_kernel<<<(MROWS * D) / 256, 256, 0, stream>>>(x, Xb);

  dim3 gg(MROWS / 128, D / 64);   // 16 x 16 blocks, 256 threads each
  gemm_bf16_kernel<<<gg, 256, 0, stream>>>(Xb, Wqd, bq, (void*)Qd,  1);
  gemm_bf16_kernel<<<gg, 256, 0, stream>>>(Xb, Wkd, bk, (void*)Kd,  1);
  gemm_bf16_kernel<<<gg, 256, 0, stream>>>(Xb, Wvd, bv, (void*)Vtd, 2);

  attn_kernel<<<(BATCH * HEADS * 64) / 4, 128, 0, stream>>>(Qd, Kd, Vtd, Aod);

  gemm_bf16_kernel<<<gg, 256, 0, stream>>>(Aod, Wod, bo, d_out, 3);
}
